// MSDeformAttn_23055384445774
// MI455X (gfx1250) — matrix operand order repaired and checked by reading, not compiled, not run on hardware
//
#include <hip/hip_runtime.h>
#include <hip/hip_bf16.h>

// ---- problem constants (match reference) ----
#define DQ   256
#define NH   8
#define NL   3
#define NP   4
#define HD   32
#define BB   8
#define NQ   2048
#define NV   21504          // 128*128 + 64*64 + 32*32

typedef __attribute__((ext_vector_type(16))) _Float16 v16h;
typedef __attribute__((ext_vector_type(8)))  float    v8f;

__device__ __forceinline__ void store_out(float* p, float v)    { *p = v; }
__device__ __forceinline__ void store_out(_Float16* p, float v) { *p = (_Float16)v; }

// ---------------------------------------------------------------------------
// Pack W[K,N] (fp32, row-major) into WMMA-B f16 layout:
//   tile (nt,kt): per lane L, 16 contiguous halves = column nt*16+(L&15),
//   K = kt*32 + (L>>4)*16 + i   (lanes 0-15 hold K 0..15, lanes 16-31 K 16..31)
// ---------------------------------------------------------------------------
__global__ void pack_b_kernel(const float* __restrict__ W, _Float16* __restrict__ Bp,
                              int K, int N) {
  const int Ktiles = K >> 5;
  const int Ntiles = N >> 4;
  const int idx = blockIdx.x * blockDim.x + threadIdx.x;
  if (idx >= Ntiles * Ktiles * 32) return;
  const int lane = idx & 31;
  const int kt   = (idx >> 5) % Ktiles;
  const int nt   = (idx >> 5) / Ktiles;
  const int col   = nt * 16 + (lane & 15);
  const int kbase = kt * 32 + (lane >> 4) * 16;
  _Float16* dst = Bp + (size_t)idx * 16;
#pragma unroll
  for (int i = 0; i < 16; ++i)
    dst[i] = (_Float16)W[(size_t)(kbase + i) * N + col];
}

// ---------------------------------------------------------------------------
// WMMA GEMM: C[M,N] = A[M,K](fp32,row-major, cvt->f16 in-reg) * Bp(packed f16)
//            + bias[N].  One wave -> 16 x (16*NT) tile.
// K is a compile-time constant -> fully unrolled K loop (KT WMMA batches),
// letting the scheduler hoist next-tile loads above the current WMMA batch.
// ---------------------------------------------------------------------------
template <int NT, int KCONST, typename OutT>
__global__ void wmma_gemm_kernel(const float* __restrict__ A,
                                 const _Float16* __restrict__ Bp,
                                 const float* __restrict__ bias,
                                 OutT* __restrict__ C,
                                 int M, int N) {
  constexpr int K      = KCONST;
  constexpr int Ktiles = K >> 5;

  const int lane  = threadIdx.x & 31;
  const int wave  = threadIdx.x >> 5;
  const int ngrp  = N / (16 * NT);
  const int gw    = blockIdx.x * (blockDim.x >> 5) + wave;
  const int mt    = gw / ngrp;
  const int ng    = gw % ngrp;
  if (mt * 16 >= M) return;               // wave-uniform: EXEC stays all-ones

  const int row   = mt * 16 + (lane & 15);
  const int khalf = (lane >> 4) * 16;     // 0 or 16

  v8f acc[NT];
#pragma unroll
  for (int t = 0; t < NT; ++t) acc[t] = {};

  const float* __restrict__ arow = A + (size_t)row * K;

#pragma unroll
  for (int kt = 0; kt < Ktiles; ++kt) {
    const int kb = kt * 32;
    // A operand: two contiguous 8-float runs -> 16 halves (k = khalf + i, same order as packed B)
    const float4 f0 = ((const float4*)(arow + kb + khalf))[0];
    const float4 f1 = ((const float4*)(arow + kb + khalf))[1];
    const float4 f2 = ((const float4*)(arow + kb + khalf + 8))[0];
    const float4 f3 = ((const float4*)(arow + kb + khalf + 8))[1];
    if (kt + 1 < Ktiles) {                // prefetch next A chunk (global_prefetch_b8)
      __builtin_prefetch(arow + kb + 32 + khalf, 0, 0);
    }
    v16h a;
    a[0]=(_Float16)f0.x;  a[1]=(_Float16)f0.y;  a[2]=(_Float16)f0.z;  a[3]=(_Float16)f0.w;
    a[4]=(_Float16)f1.x;  a[5]=(_Float16)f1.y;  a[6]=(_Float16)f1.z;  a[7]=(_Float16)f1.w;
    a[8]=(_Float16)f2.x;  a[9]=(_Float16)f2.y;  a[10]=(_Float16)f2.z; a[11]=(_Float16)f2.w;
    a[12]=(_Float16)f3.x; a[13]=(_Float16)f3.y; a[14]=(_Float16)f3.z; a[15]=(_Float16)f3.w;

#pragma unroll
    for (int t = 0; t < NT; ++t) {
      const int nt = ng * NT + t;
      const v16h b = *(const v16h*)(Bp + ((((size_t)nt * Ktiles + kt) * 32 + lane) * 16));
      acc[t] = __builtin_amdgcn_wmma_f32_16x16x32_f16(
          false, a, false, b, (short)0, acc[t], false, false);
    }
  }

  // epilogue: C/D layout — VGPR r: lanes 0-15 -> M=r, lanes 16-31 -> M=8+r
  const int crow = mt * 16 + (lane >> 4) * 8;
  const int cc   = lane & 15;
#pragma unroll
  for (int t = 0; t < NT; ++t) {
    const int col = (ng * NT + t) * 16 + cc;
    const float bv = bias[col];
#pragma unroll
    for (int r = 0; r < 8; ++r)
      store_out(C + (size_t)(crow + r) * N + col, acc[t][r] + bv);
  }
}

// ---------------------------------------------------------------------------
// Softmax over NL*NP = 12 logits per (b,q,head), in place.
// ---------------------------------------------------------------------------
__global__ void softmax12_kernel(float* __restrict__ attn, int total) {
  const int i = blockIdx.x * blockDim.x + threadIdx.x;
  if (i >= total) return;
  float* p = attn + (size_t)(i >> 3) * (NH * NL * NP) + (i & 7) * (NL * NP);
  float m = p[0];
#pragma unroll
  for (int k = 1; k < 12; ++k) m = fmaxf(m, p[k]);
  float s = 0.0f;
  float e[12];
#pragma unroll
  for (int k = 0; k < 12; ++k) { e[k] = __expf(p[k] - m); s += e[k]; }
  const float inv = 1.0f / s;
#pragma unroll
  for (int k = 0; k < 12; ++k) p[k] = e[k] * inv;
}

// ---------------------------------------------------------------------------
// Bilinear sampling + attention-weighted sum.
// One wave per (b,q,head); lane = channel d in [0,32).  Value tensor in f16.
// Corner coords are identical across the wave -> branches are wave-uniform.
// ---------------------------------------------------------------------------
__device__ __forceinline__ float corner_tap(const _Float16* __restrict__ v,
                                            int xi, int yi, int W, int H, float w) {
  if (xi < 0 || xi >= W || yi < 0 || yi >= H) return 0.0f;
  return w * (float)v[(size_t)(yi * W + xi) * DQ];
}

__global__ void ms_sample_kernel(const float* __restrict__ ref,     // [B,NQ,NL,2]
                                 const float* __restrict__ off,     // [B*NQ,192]
                                 const float* __restrict__ attn,    // [B*NQ,96]
                                 const _Float16* __restrict__ vproj,// [B,NV,256] f16
                                 float* __restrict__ out) {         // [B*NQ,256]
  const int gw   = blockIdx.x * (blockDim.x >> 5) + (threadIdx.x >> 5);
  const int lane = threadIdx.x & 31;
  const int h    = gw & 7;
  const int bq   = gw >> 3;
  if (bq >= BB * NQ) return;
  const int b = bq / NQ;

  const int Hs[NL] = {128, 64, 32};
  const int Ws[NL] = {128, 64, 32};
  const int Ss[NL] = {0, 16384, 20480};

  const float* __restrict__ offp  = off  + (size_t)bq * (NH * NL * NP * 2) + h * (NL * NP * 2);
  const float* __restrict__ attnp = attn + (size_t)bq * (NH * NL * NP)     + h * (NL * NP);
  const float* __restrict__ refp  = ref  + (size_t)bq * (NL * 2);
  const _Float16* __restrict__ vb = vproj + (size_t)b * NV * DQ + h * HD + lane;

  float acc = 0.0f;
#pragma unroll
  for (int l = 0; l < NL; ++l) {
    const float rx = refp[l * 2 + 0];
    const float ry = refp[l * 2 + 1];
    const int   Hl = Hs[l], Wl = Ws[l];
    const _Float16* __restrict__ vl = vb + (size_t)Ss[l] * DQ;
#pragma unroll
    for (int p = 0; p < NP; ++p) {
      const float ox = offp[l * (NP * 2) + p * 2 + 0];
      const float oy = offp[l * (NP * 2) + p * 2 + 1];
      // loc = ref + off/(W,H); grid=2*loc-1; x=(grid+1)*0.5*W-0.5 == loc*W-0.5
      const float x = (rx + ox / (float)Wl) * (float)Wl - 0.5f;
      const float y = (ry + oy / (float)Hl) * (float)Hl - 0.5f;
      const float x0f = floorf(x), y0f = floorf(y);
      const float wx = x - x0f, wy = y - y0f;
      const int   x0 = (int)x0f, y0 = (int)y0f;
      const float aw = attnp[l * NP + p];
      acc += corner_tap(vl, x0,     y0,     Wl, Hl, aw * (1.0f - wx) * (1.0f - wy));
      acc += corner_tap(vl, x0 + 1, y0,     Wl, Hl, aw * wx * (1.0f - wy));
      acc += corner_tap(vl, x0,     y0 + 1, Wl, Hl, aw * (1.0f - wx) * wy);
      acc += corner_tap(vl, x0 + 1, y0 + 1, Wl, Hl, aw * wx * wy);
    }
  }
  out[(size_t)bq * DQ + h * HD + lane] = acc;
}

// ---------------------------------------------------------------------------
extern "C" void kernel_launch(void* const* d_in, const int* in_sizes, int n_in,
                              void* d_out, int out_size, void* d_ws, size_t ws_size,
                              hipStream_t stream) {
  const float* query = (const float*)d_in[0];   // [B,NQ,256]
  const float* refpt = (const float*)d_in[1];   // [B,NQ,NL,2]
  const float* value = (const float*)d_in[2];   // [B,NV,256]
  const float* W_off = (const float*)d_in[5];   // [256,192]
  const float* b_off = (const float*)d_in[6];
  const float* W_attn= (const float*)d_in[7];   // [256,96]
  const float* b_attn= (const float*)d_in[8];
  const float* W_val = (const float*)d_in[9];   // [256,256]
  const float* b_val = (const float*)d_in[10];
  const float* W_out = (const float*)d_in[11];  // [256,256]
  const float* b_out = (const float*)d_in[12];
  float* out = (float*)d_out;

  // ---- workspace partition (256B aligned) ----
  char* ws = (char*)d_ws;
  size_t o = 0;
  auto take = [&](size_t bytes) { char* p = ws + o; o += (bytes + 255) & ~(size_t)255; return p; };
  _Float16* BpVal  = (_Float16*)take((size_t)DQ * DQ  * 2);
  _Float16* BpOff  = (_Float16*)take((size_t)DQ * 192 * 2);
  _Float16* BpAttn = (_Float16*)take((size_t)DQ * 96  * 2);
  _Float16* BpOut  = (_Float16*)take((size_t)DQ * DQ  * 2);
  _Float16* vprojH = (_Float16*)take((size_t)BB * NV * DQ * 2);
  float*    offF   = (float*)   take((size_t)BB * NQ * 192 * 4);
  float*    attnF  = (float*)   take((size_t)BB * NQ * 96  * 4);
  float*    sampF  = (float*)   take((size_t)BB * NQ * DQ  * 4);
  (void)ws_size; (void)in_sizes; (void)n_in; (void)out_size;

  // ---- 1) pack weights into WMMA-B layout ----
  {
    int t;
    t = (DQ / 16) * (DQ / 32) * 32;   // 4096
    pack_b_kernel<<<(t + 255) / 256, 256, 0, stream>>>(W_val,  BpVal,  DQ, DQ);
    pack_b_kernel<<<(t + 255) / 256, 256, 0, stream>>>(W_out,  BpOut,  DQ, DQ);
    t = (192 / 16) * (DQ / 32) * 32;  // 3072
    pack_b_kernel<<<(t + 255) / 256, 256, 0, stream>>>(W_off,  BpOff,  DQ, 192);
    t = (96 / 16) * (DQ / 32) * 32;   // 1536
    pack_b_kernel<<<(t + 255) / 256, 256, 0, stream>>>(W_attn, BpAttn, DQ, 96);
  }

  // ---- 2) value projection -> f16   [B*NV,256] ----
  {
    const int M = BB * NV, N = DQ;
    const int waves = (M / 16) * (N / 64);          // NT=4
    wmma_gemm_kernel<4, DQ, _Float16><<<(waves + 3) / 4, 128, 0, stream>>>(
        value, BpVal, b_val, vprojH, M, N);
  }

  // ---- 3) sampling offsets  [B*NQ,192] ----
  {
    const int M = BB * NQ, N = 192;
    const int waves = (M / 16) * (N / 64);          // NT=4
    wmma_gemm_kernel<4, DQ, float><<<(waves + 3) / 4, 128, 0, stream>>>(
        query, BpOff, b_off, offF, M, N);
  }

  // ---- 4) attention logits  [B*NQ,96] ----
  {
    const int M = BB * NQ, N = 96;
    const int waves = (M / 16) * (N / 32);          // NT=2
    wmma_gemm_kernel<2, DQ, float><<<(waves + 3) / 4, 128, 0, stream>>>(
        query, BpAttn, b_attn, attnF, M, N);
  }

  // ---- 5) softmax over 12 (level,point) logits per (b,q,head) ----
  {
    const int total = BB * NQ * NH;
    softmax12_kernel<<<(total + 255) / 256, 256, 0, stream>>>(attnF, total);
  }

  // ---- 6) bilinear sampling + weighted sum -> [B*NQ,256] ----
  {
    const int waves = BB * NQ * NH;                 // one wave per (b,q,h)
    ms_sample_kernel<<<(waves + 7) / 8, 256, 0, stream>>>(
        refpt, offF, attnF, vprojH, sampF);
  }

  // ---- 7) output projection -> d_out  [B*NQ,256] ----
  {
    const int M = BB * NQ, N = DQ;
    const int waves = (M / 16) * (N / 64);          // NT=4
    wmma_gemm_kernel<4, DQ, float><<<(waves + 3) / 4, 128, 0, stream>>>(
        sampF, BpOut, b_out, out, M, N);
  }
}
